// CrossAttention_43327630082762
// MI455X (gfx1250) — compile-verified, matrix operand order checked by reading, not run on hardware
//
#include <hip/hip_runtime.h>
#include <hip/hip_bf16.h>

// ---------------------------------------------------------------------------
// Fused shifted-window cross-attention for MI455X (gfx1250), wave32 + WMMA.
// Prologue kernel converts weights fp32 -> bf16 pre-transposed into fragment
// order (WT[n][k]) in d_ws.  Main kernel: one workgroup per 7x7 window
// (padded to 64 tokens); Q,K,V,P,O live in LDS as bf16; all matmuls are
// v_wmma_f32_16x16x32_bf16 with f32 accumulation.  B fragments stream
// directly from L2 as global_load_b128 (no LDS staging, no extra barriers);
// A fragments are ds_load_b128 pairs from AS3 pointers.
// ---------------------------------------------------------------------------

#define B_    32
#define HH    56
#define WW_   56
#define C_    256
#define NH_   8
#define HD_   32
#define WIN_  7
#define SH_   3
#define WA_   49
#define WAP_  64   // padded tokens
#define NWX_  8    // windows per row/col

#define AS3 __attribute__((address_space(3)))
#define AS1 __attribute__((address_space(1)))

typedef __attribute__((ext_vector_type(16))) __bf16        bf16x16;
typedef __attribute__((ext_vector_type(8)))  float         f32x8;
typedef __attribute__((ext_vector_type(4)))  unsigned int  u32x4;   // POD 16B vector

__device__ __forceinline__ unsigned short f2bf(float x) {
    union { __bf16 b; unsigned short u; } c;
    c.b = (__bf16)x;                       // hardware RNE f32->bf16 convert
    return c.u;
}

// 16x32 / 32x16 bf16 WMMA fragment: per the CDNA5 ISA layout each lane holds
// K = k0..k0+7 and K = 16+k0..16+k0+7 (k0 = 8*(lane>>4)) -> two 16B loads.
__device__ __forceinline__ bf16x16 load_frag(const AS3 unsigned short* lo,
                                             const AS3 unsigned short* hi) {
    union { bf16x16 v; u32x4 q[2]; } u;
    u.q[0] = *(const AS3 u32x4*)lo;
    u.q[1] = *(const AS3 u32x4*)hi;
    return u.v;
}

__device__ __forceinline__ bf16x16 load_frag_g(const unsigned short* lo,
                                               const unsigned short* hi) {
    union { bf16x16 v; u32x4 q[2]; } u;
    u.q[0] = *(const AS1 u32x4*)lo;        // global_load_b128
    u.q[1] = *(const AS1 u32x4*)hi;
    return u.v;
}

__device__ __forceinline__ int region_of(int p) {      // shift-mask region id
    return p < (HH - WIN_) ? 0 : (p < (HH - SH_) ? 1 : 2);
}

// ---- Prologue: weights fp32 -> bf16, transposed to WT[m][n][k] ----
extern "C" __global__ __launch_bounds__(256, 8)
void weight_prep_kernel(const float* __restrict__ wq, const float* __restrict__ wkv,
                        const float* __restrict__ wo, unsigned short* __restrict__ wT) {
    int idx = blockIdx.x * 256 + threadIdx.x;   // 0 .. 4*256*256-1
    int m   = idx >> 16;
    int rem = idx & 65535;
    int n   = rem >> 8, k = rem & 255;
    float v;
    if      (m == 0) v = wq [k * 256 + n];            // Wq
    else if (m == 1) v = wkv[k * 512 + n];            // Wk  (cols   0..255)
    else if (m == 2) v = wkv[k * 512 + 256 + n];      // Wv  (cols 256..511)
    else             v = wo [k * 256 + n];            // Wo
    wT[idx] = f2bf(v);
}

// GEMM: [64 x 256](bf16, LDS) @ WT^T ([256 x 256] bf16, global, row n holds
// column n of the weight matrix) + bias.  mode 0: bf16 row-major to sDst
// (ld 256). mode 1: bf16 transposed to sDst (ld 64). mode 2: f32 scatter.
__device__ void gemm_tokens(const AS3 unsigned short* sA,
                            const unsigned short* __restrict__ WT,
                            const float* bias, int mode, AS3 unsigned short* sDst,
                            float* gOut, int bIdx, int wh, int ww, int tid) {
    const int wave = tid >> 5, lane = tid & 31;
    const int hl = lane >> 4, ln = lane & 15;
    const int k0 = hl * 8;

    f32x8 acc[4][2];
#pragma unroll
    for (int mt = 0; mt < 4; ++mt)
#pragma unroll
        for (int nt = 0; nt < 2; ++nt) acc[mt][nt] = (f32x8){};

#pragma unroll
    for (int kb = 0; kb < 8; ++kb) {
        bf16x16 bfr[2];
#pragma unroll
        for (int nt = 0; nt < 2; ++nt) {
            int col = (wave * 2 + nt) * 16 + ln;
            bfr[nt] = load_frag_g(WT + col * 256 + kb * 32 + k0,
                                  WT + col * 256 + kb * 32 + 16 + k0);
        }
#pragma unroll
        for (int mt = 0; mt < 4; ++mt) {
            int row = mt * 16 + ln;
            bf16x16 a = load_frag(sA + row * 256 + kb * 32 + k0,
                                  sA + row * 256 + kb * 32 + 16 + k0);
#pragma unroll
            for (int nt = 0; nt < 2; ++nt)
                acc[mt][nt] = __builtin_amdgcn_wmma_f32_16x16x32_bf16(
                    false, a, false, bfr[nt], (short)0, acc[mt][nt], false, false);
        }
    }

#pragma unroll
    for (int mt = 0; mt < 4; ++mt)
#pragma unroll
        for (int nt = 0; nt < 2; ++nt) {
            int col = (wave * 2 + nt) * 16 + ln;
            float bv = bias[col];
#pragma unroll
            for (int r = 0; r < 8; ++r) {
                int m = mt * 16 + r + hl * 8;
                float v = acc[mt][nt][r] + bv;
                if (mode == 0) {
                    sDst[m * 256 + col] = f2bf(v);
                } else if (mode == 1) {
                    sDst[col * 64 + m] = f2bf(v);
                } else {
                    if (m < WA_) {
                        int i = m / 7, j = m % 7;
                        int h0 = (wh * 7 + i + SH_) % HH;
                        int w0 = (ww * 7 + j + SH_) % WW_;
                        gOut[((size_t)bIdx * (HH * WW_) + h0 * WW_ + w0) * C_ + col] = v;
                    }
                }
            }
        }
    __syncthreads();
}

extern "C" __global__ __launch_bounds__(256, 1)
void swin_xattn_kernel(const float* __restrict__ x, const float* __restrict__ ctx,
                       const unsigned short* __restrict__ wT,
                       const float* __restrict__ bq, const float* __restrict__ bkv,
                       const float* __restrict__ bo, const float* __restrict__ relt,
                       float* __restrict__ out) {
    extern __shared__ unsigned char smem[];
    AS3 unsigned short* sX  = (AS3 unsigned short*)smem;  // [64][256]
    AS3 unsigned short* sC  = sX  + 64 * 256;             // [64][256]
    AS3 unsigned short* sQ  = sC  + 64 * 256;             // [64][256]
    AS3 unsigned short* sK  = sQ  + 64 * 256;             // [64][256]
    AS3 unsigned short* sVt = sK  + 64 * 256;             // [256][64] (transposed V)
    AS3 unsigned short* sP  = sVt + 256 * 64;             // [8][64][64] probs
    AS3 unsigned short* sO  = sP  + 8 * 64 * 64;          // [64][256]

    const int tid  = threadIdx.x;
    const int wb   = blockIdx.x;          // window-batch in [0, B*64)
    const int bIdx = wb >> 6;
    const int wIdx = wb & 63;
    const int wh   = wIdx >> 3, ww = wIdx & 7;

    const int wave = tid >> 5, lane = tid & 31;
    const int hl = lane >> 4, ln = lane & 15;
    const int k0 = hl * 8;

    // ---- Phase 0: gather rolled windows of x and context into LDS (bf16) ----
    const float* xg = x   + (size_t)bIdx * (HH * WW_) * C_;
    const float* cg = ctx + (size_t)bIdx * (HH * WW_) * C_;
#pragma unroll
    for (int idx = tid; idx < WA_ * 256; idx += 256) {
        int t = idx >> 8, ch = idx & 255;
        int i = t / 7, j = t % 7;
        int h0 = (wh * 7 + i + SH_) % HH;
        int w0 = (ww * 7 + j + SH_) % WW_;
        size_t off = (size_t)(h0 * WW_ + w0) * C_ + ch;
        sX[t * 256 + ch] = f2bf(xg[off]);
        sC[t * 256 + ch] = f2bf(cg[off]);
    }
#pragma unroll
    for (int idx = tid; idx < (WAP_ - WA_) * 256; idx += 256) {
        int t = WA_ + (idx >> 8), ch = idx & 255;
        sX[t * 256 + ch] = 0;
        sC[t * 256 + ch] = 0;
    }
    __syncthreads();

    // ---- Phase 1: projections (bf16 WMMA, f32 accumulate) ----
    const unsigned short* wqT = wT;
    const unsigned short* wkT = wT + 1 * 65536;
    const unsigned short* wvT = wT + 2 * 65536;
    const unsigned short* woT = wT + 3 * 65536;
    gemm_tokens(sX, wqT, bq,        0, sQ,  nullptr, 0, 0, 0, tid);  // Q
    gemm_tokens(sC, wkT, bkv,       0, sK,  nullptr, 0, 0, 0, tid);  // K
    gemm_tokens(sC, wvT, bkv + 256, 1, sVt, nullptr, 0, 0, 0, tid);  // V^T

    // ---- Phase 2: per-head attention (wave = head) ----
    const int h = wave;
    const float scale = 0.17677669529663687f;  // 1/sqrt(32)

#pragma unroll
    for (int mt = 0; mt < 4; ++mt) {
        f32x8 s[4];
        int qrow = mt * 16 + ln;
        bf16x16 qa = load_frag(sQ + qrow * 256 + h * 32 + k0,
                               sQ + qrow * 256 + h * 32 + 16 + k0);
#pragma unroll
        for (int nt = 0; nt < 4; ++nt) {
            int krow = nt * 16 + ln;
            bf16x16 kf = load_frag(sK + krow * 256 + h * 32 + k0,
                                   sK + krow * 256 + h * 32 + 16 + k0);
            f32x8 z = (f32x8){};
            s[nt] = __builtin_amdgcn_wmma_f32_16x16x32_bf16(
                false, qa, false, kf, (short)0, z, false, false);
        }
        // fused bias + shift-mask + softmax (row = 16 lanes of a half-wave)
#pragma unroll
        for (int r = 0; r < 8; ++r) {
            int m = mt * 16 + r + hl * 8;
            bool vq = m < WA_;
            int qi = m / 7, qj = m % 7;
            int rq = region_of(wh * 7 + qi) * 3 + region_of(ww * 7 + qj);
            float vals[4];
            float mx = -1e30f;
#pragma unroll
            for (int nt = 0; nt < 4; ++nt) {
                int n = nt * 16 + ln;
                float sv = -1e30f;
                if (vq && n < WA_) {
                    int ki = n / 7, kj = n % 7;
                    int rk = region_of(wh * 7 + ki) * 3 + region_of(ww * 7 + kj);
                    float bias = relt[((ki - qi + 6) * 13 + (kj - qj + 6)) * NH_ + h];
                    sv = s[nt][r] * scale + bias + (rk == rq ? 0.0f : -100.0f);
                }
                vals[nt] = sv;
                mx = fmaxf(mx, sv);
            }
#pragma unroll
            for (int off = 1; off < 16; off <<= 1)
                mx = fmaxf(mx, __shfl_xor(mx, off, 32));
            float sum = 0.0f;
#pragma unroll
            for (int nt = 0; nt < 4; ++nt) {
                vals[nt] = __expf(vals[nt] - mx);
                sum += vals[nt];
            }
#pragma unroll
            for (int off = 1; off < 16; off <<= 1)
                sum += __shfl_xor(sum, off, 32);
            float inv = vq ? (1.0f / sum) : 0.0f;
#pragma unroll
            for (int nt = 0; nt < 4; ++nt)
                sP[h * 4096 + m * 64 + nt * 16 + ln] = f2bf(vals[nt] * inv);
        }
    }

    // P @ V_head  ->  sO columns [h*32, h*32+32)
#pragma unroll
    for (int mt = 0; mt < 4; ++mt)
#pragma unroll
        for (int nv = 0; nv < 2; ++nv) {
            f32x8 o = (f32x8){};
#pragma unroll
            for (int kc = 0; kc < 2; ++kc) {
                int prow = mt * 16 + ln;
                bf16x16 pa = load_frag(sP + h * 4096 + prow * 64 + kc * 32 + k0,
                                       sP + h * 4096 + prow * 64 + kc * 32 + 16 + k0);
                int vcol = h * 32 + nv * 16 + ln;
                bf16x16 vb = load_frag(sVt + vcol * 64 + kc * 32 + k0,
                                       sVt + vcol * 64 + kc * 32 + 16 + k0);
                o = __builtin_amdgcn_wmma_f32_16x16x32_bf16(
                    false, pa, false, vb, (short)0, o, false, false);
            }
#pragma unroll
            for (int r = 0; r < 8; ++r) {
                int m = mt * 16 + r + hl * 8;
                sO[m * 256 + h * 32 + nv * 16 + ln] = f2bf(o[r]);
            }
        }
    __syncthreads();

    // ---- Phase 3: output projection + inverse roll scatter to global ----
    gemm_tokens(sO, woT, bo, 2, sO, out, bIdx, wh, ww, tid);
}

#define SMEM_BYTES ((6 * 64 * 256 + 8 * 64 * 64) * 2)   // 256 KB

extern "C" void kernel_launch(void* const* d_in, const int* in_sizes, int n_in,
                              void* d_out, int out_size, void* d_ws, size_t ws_size,
                              hipStream_t stream) {
    (void)in_sizes; (void)n_in; (void)out_size; (void)ws_size;
    const float* x    = (const float*)d_in[0];
    const float* ctx  = (const float*)d_in[1];
    const float* wq   = (const float*)d_in[2];
    const float* bq   = (const float*)d_in[3];
    const float* wkv  = (const float*)d_in[4];
    const float* bkv  = (const float*)d_in[5];
    const float* wo   = (const float*)d_in[6];
    const float* bo   = (const float*)d_in[7];
    const float* relt = (const float*)d_in[8];
    float* out = (float*)d_out;
    unsigned short* wT = (unsigned short*)d_ws;   // 4 * 256*256 bf16 = 512 KB

    // Prologue: convert + transpose all weights into fragment order (once per call).
    weight_prep_kernel<<<dim3(4 * 256 * 256 / 256), dim3(256), 0, stream>>>(wq, wkv, wo, wT);

    (void)hipFuncSetAttribute((const void*)swin_xattn_kernel,
                              hipFuncAttributeMaxDynamicSharedMemorySize, SMEM_BYTES);

    dim3 grid(B_ * NWX_ * NWX_);   // 2048 windows
    dim3 block(256);               // 8 waves; wave == head in attention phase
    swin_xattn_kernel<<<grid, block, SMEM_BYTES, stream>>>(
        x, ctx, wT, bq, bkv, bo, relt, out);
}
